// MergedColumnParallelLinearWithTopping_69638599737693
// MI455X (gfx1250) — matrix operand order repaired and checked by reading, not compiled, not run on hardware
//
#include <hip/hip_runtime.h>

// ---------------------------------------------------------------------------
// MergedColumnParallelLinearWithTopping for MI455X (gfx1250, wave32, WMMA)
//   out[t,n] = sum_d x[t,d]*W[n,d] + sum_r xa[t,16*half(n)+r]*B_buffer[idx[t],r,n]
//   xa[t,r]  = sum_d x[t,d]*A_buffer[idx[t],d,r]          (r in 0..31)
//
// Compute-bound at bf16 matrix-core rate (94.5 GFLOP GEMM vs 8.6us memory
// floor at 23.3 TB/s): convert x/W to bf16 once (operands then L2-resident:
// 46MB W + 8MB x << 192MB L2), run v_wmma_f32_16x16x32_bf16.
// Wave tile 32x64 (2 A-frags x 4 B-frags -> 8 wmma, 12 b128 loads per k-step,
// 1.5 loads/wmma), block tile 64x256, LoRA stage-2 fused in the epilogue
// (B_buffer 11.5MB -> L2-resident). global_prefetch covers L2 latency.
// ---------------------------------------------------------------------------

#define T_TOK   2048
#define D_DIM   2048
#define N_OUT   11264          // 2*BDIM
#define B_HALF  5632           // BDIM
#define R_RANK  16

typedef __attribute__((ext_vector_type(16))) __bf16 v16bf;
typedef __attribute__((ext_vector_type(8)))  float  v8f;

union BF16x16 { v16bf v; uint4 q[2]; };

// ---- fp32 -> bf16 round-to-nearest-even -----------------------------------
static __device__ __forceinline__ unsigned short f2bf(float f) {
    union { float f; unsigned u; } c; c.f = f;
    unsigned u = c.u;
    u = (u + 0x7FFFu + ((u >> 16) & 1u)) >> 16;  // RNE
    return (unsigned short)u;
}

__global__ __launch_bounds__(256)
void cvt_bf16_kernel(const float* __restrict__ src,
                     unsigned short* __restrict__ dst, int n4) {
    int i = blockIdx.x * 256 + threadIdx.x;
    if (i >= n4) return;
    float4 f = ((const float4*)src)[i];
    ushort4 o;
    o.x = f2bf(f.x); o.y = f2bf(f.y); o.z = f2bf(f.z); o.w = f2bf(f.w);
    ((ushort4*)dst)[i] = o;
}

// ---- stage 1 LoRA: xa[t, r] = sum_d x[t,d] * A[idx[t], d, r], r = lane ----
__global__ __launch_bounds__(256)
void lora_xa_kernel(const float* __restrict__ x,
                    const float* __restrict__ A,
                    const int*   __restrict__ widx,
                    float*       __restrict__ xa) {
    const int lane = threadIdx.x & 31;
    const int wave = threadIdx.x >> 5;
    const int t    = blockIdx.x * 8 + wave;
    const int idx  = widx[t];
    const float* xp = x + (size_t)t * D_DIM;
    const float* Ap = A + (size_t)idx * (size_t)(D_DIM * 2 * R_RANK) + lane;
    float acc = 0.f;
#pragma unroll 8
    for (int d = 0; d < D_DIM; ++d)
        acc = fmaf(xp[d], Ap[(size_t)d * (2 * R_RANK)], acc);
    xa[t * (2 * R_RANK) + lane] = acc;
}

// ---- main GEMM + fused LoRA stage 2 ---------------------------------------
__global__ __launch_bounds__(256)
void gemm_lora_kernel(const unsigned short* __restrict__ xb,   // T x D bf16
                      const unsigned short* __restrict__ Wb,   // N x D bf16
                      const float*          __restrict__ Bbuf, // L x R x N f32
                      const float*          __restrict__ xa,   // T x 32 f32
                      const int*            __restrict__ widx, // T
                      float*                __restrict__ out)  // T x N f32
{
    const int lane = threadIdx.x & 31;
    const int wave = threadIdx.x >> 5;
    const int h    = lane >> 4;    // half-wave (0/1)
    const int ln   = lane & 15;
    const int wm   = wave & 1;     // 2 waves along M (32 rows each)
    const int wn   = wave >> 1;    // 4 waves along N (64 cols each)
    const int m0   = blockIdx.y * 64  + wm * 32;
    const int n0   = blockIdx.x * 256 + wn * 64;

    // A-frag rows (lanes 0-15 and 16-31 both cover M=0..15 of each sub-tile)
    const unsigned short* xrow0 = xb + (size_t)(m0 + ln) * D_DIM;
    const unsigned short* xrow1 = xrow0 + (size_t)16 * D_DIM;

    v8f acc[2][4] = {};   // 2(M) x 4(N) tiles of 16x16 f32 -> wave tile 32x64

    for (int k0 = 0; k0 < D_DIM; k0 += 32) {
        // A 16x32 bf16 frags: per-lane chunks at K = k0+8h and K = k0+16+8h
        BF16x16 a0, a1;
        a0.q[0] = *(const uint4*)(xrow0 + k0 + 8 * h);
        a0.q[1] = *(const uint4*)(xrow0 + k0 + 16 + 8 * h);
        a1.q[0] = *(const uint4*)(xrow1 + k0 + 8 * h);
        a1.q[1] = *(const uint4*)(xrow1 + k0 + 16 + 8 * h);
        __builtin_prefetch(xrow0 + k0 + 128, 0, 3);   // 4 k-steps ahead
        __builtin_prefetch(xrow1 + k0 + 128, 0, 3);
#pragma unroll
        for (int j = 0; j < 4; ++j) {
            // B 32x16 frag: lane = col n, W row n chunks at K = k0+8h and K = k0+16+8h
            const unsigned short* wrow =
                Wb + (size_t)(n0 + j * 16 + ln) * D_DIM + k0 + 8 * h;
            BF16x16 b;
            b.q[0] = ((const uint4*)wrow)[0];
            b.q[1] = ((const uint4*)wrow)[2];
            __builtin_prefetch(wrow + 128, 0, 3);
            acc[0][j] = __builtin_amdgcn_wmma_f32_16x16x32_bf16(
                false, a0.v, false, b.v, (short)0, acc[0][j], false, false);
            acc[1][j] = __builtin_amdgcn_wmma_f32_16x16x32_bf16(
                false, a1.v, false, b.v, (short)0, acc[1][j], false, false);
        }
    }

    // Epilogue: fused LoRA stage 2. 5632 = 22*256, so the 256-wide block tile
    // never straddles the half boundary -> halfI is block-uniform; B_buffer is
    // indexed with the absolute output column.
    const int halfI = n0 / B_HALF;
#pragma unroll
    for (int mi = 0; mi < 2; ++mi) {
#pragma unroll
        for (int v = 0; v < 8; ++v) {
            const int t   = m0 + mi * 16 + v + 8 * h;  // C layout: row = v + 8h
            const int idx = widx[t];
            const float4* xap4 = (const float4*)(xa + t * (2 * R_RANK) + halfI * R_RANK);
            const float4 x0 = xap4[0], x1 = xap4[1], x2 = xap4[2], x3 = xap4[3];
            const float xv[16] = { x0.x, x0.y, x0.z, x0.w,  x1.x, x1.y, x1.z, x1.w,
                                   x2.x, x2.y, x2.z, x2.w,  x3.x, x3.y, x3.z, x3.w };
            const float* bbase = Bbuf + (size_t)idx * (size_t)(R_RANK * N_OUT);
#pragma unroll
            for (int j = 0; j < 4; ++j) {
                const int n_abs = n0 + j * 16 + ln;
                const float* bp = bbase + n_abs;
                float dsum = 0.f;
#pragma unroll
                for (int r = 0; r < R_RANK; ++r)
                    dsum = fmaf(xv[r], bp[(size_t)r * N_OUT], dsum);
                out[(size_t)t * N_OUT + n_abs] = acc[mi][j][v] + dsum;
            }
        }
    }
}

// ---------------------------------------------------------------------------
extern "C" void kernel_launch(void* const* d_in, const int* in_sizes, int n_in,
                              void* d_out, int out_size, void* d_ws, size_t ws_size,
                              hipStream_t stream) {
    const float* x    = (const float*)d_in[0];
    const float* W    = (const float*)d_in[1];
    const float* A    = (const float*)d_in[2];
    const float* Bbuf = (const float*)d_in[3];
    const int*   widx = (const int*)d_in[4];
    float*       out  = (float*)d_out;

    // workspace layout (16B aligned): xb (8MB) | Wb (46MB) | xa (256KB)
    unsigned short* xb = (unsigned short*)d_ws;
    unsigned short* Wb = xb + (size_t)T_TOK * D_DIM;
    float*          xa = (float*)(Wb + (size_t)N_OUT * D_DIM);

    {   // fp32 -> bf16 conversions (one pass; bf16 operands then L2-resident)
        int n4 = (T_TOK * D_DIM) / 4;
        cvt_bf16_kernel<<<n4 / 256, 256, 0, stream>>>(x, xb, n4);
    }
    {
        int n4 = (N_OUT * D_DIM) / 4;
        cvt_bf16_kernel<<<n4 / 256, 256, 0, stream>>>(W, Wb, n4);
    }

    // xa = einsum('td,tdr->tr') with per-token gathered A
    lora_xa_kernel<<<T_TOK / 8, 256, 0, stream>>>(x, A, widx, xa);

    // WMMA GEMM + fused LoRA delta; grid covers 11264 x 2048 exactly
    gemm_lora_kernel<<<dim3(N_OUT / 256, T_TOK / 64), 256, 0, stream>>>(
        xb, Wb, Bbuf, xa, widx, out);
}
